// MultiheadAttention_82884278878922
// MI455X (gfx1250) — matrix operand order repaired and checked by reading, not compiled, not run on hardware
//
#include <hip/hip_runtime.h>
#include <cmath>
#include <stdint.h>

// ---------------------------------------------------------------------------
// CDNA5 (gfx1250) bf16 WMMA attention pipeline.
// wave32, 256-thread blocks (8 waves), WMMA 16x16x32 bf16 -> f32,
// async global->LDS DMA (ASYNCcnt) with double-buffered software pipeline.
// ---------------------------------------------------------------------------

typedef __attribute__((ext_vector_type(8)))  __bf16 v8bf;
typedef __attribute__((ext_vector_type(16))) __bf16 v16bf;
typedef __attribute__((ext_vector_type(8)))  float  v8f;

#define BF_CAT(lo, hi) __builtin_shufflevector((lo), (hi), 0,1,2,3,4,5,6,7,8,9,10,11,12,13,14,15)
#define WMMA_BF16(A, Bm, C) \
  __builtin_amdgcn_wmma_f32_16x16x32_bf16(false, (A), false, (Bm), (short)0, (C), false, false)

// CDNA5 async global->LDS (tracked by ASYNCcnt). VDST = LDS byte address
// (low 32 bits of the flat pointer; LDS aperture lives in bits [63:32]).
__device__ __forceinline__ void async_load_b128(unsigned lds_off, const void* gaddr) {
  asm volatile("global_load_async_to_lds_b128 %0, %1, off"
               :
               : "v"(lds_off), "v"((unsigned long long)(uintptr_t)gaddr)
               : "memory");
}
__device__ __forceinline__ void wait_async0() {
  asm volatile("s_wait_asynccnt 0" ::: "memory");
}
__device__ __forceinline__ unsigned lds_off(const void* p) {
  return (unsigned)(uintptr_t)p;
}

// round-to-nearest-even f32 -> bf16 (bit-level; no scalar __bf16 conversions)
__device__ __forceinline__ unsigned short f2bf(float f) {
  unsigned int u = __float_as_uint(f);
  u += 0x7FFFu + ((u >> 16) & 1u);
  return (unsigned short)(u >> 16);
}

// butterfly reductions across the 16 lanes of a half-wave (ds_swizzle XOR)
__device__ __forceinline__ float red_max16(float v) {
  v = fmaxf(v, __int_as_float(__builtin_amdgcn_ds_swizzle(__float_as_int(v), 0x041F)));
  v = fmaxf(v, __int_as_float(__builtin_amdgcn_ds_swizzle(__float_as_int(v), 0x081F)));
  v = fmaxf(v, __int_as_float(__builtin_amdgcn_ds_swizzle(__float_as_int(v), 0x101F)));
  v = fmaxf(v, __int_as_float(__builtin_amdgcn_ds_swizzle(__float_as_int(v), 0x201F)));
  return v;
}
__device__ __forceinline__ float red_sum16(float v) {
  v += __int_as_float(__builtin_amdgcn_ds_swizzle(__float_as_int(v), 0x041F));
  v += __int_as_float(__builtin_amdgcn_ds_swizzle(__float_as_int(v), 0x081F));
  v += __int_as_float(__builtin_amdgcn_ds_swizzle(__float_as_int(v), 0x101F));
  v += __int_as_float(__builtin_amdgcn_ds_swizzle(__float_as_int(v), 0x201F));
  return v;
}

// ---------------------------------------------------------------------------
// f32 -> bf16 conversion (vectorized, grid-stride)
// ---------------------------------------------------------------------------
__global__ __launch_bounds__(256)
void cvt_f32_bf16(const float4* __restrict__ in, ushort4* __restrict__ out, int n4) {
  int i = blockIdx.x * 256 + threadIdx.x;
  const int stride = gridDim.x * 256;
  for (; i < n4; i += stride) {
    float4 f = in[i];
    ushort4 u;
    u.x = f2bf(f.x); u.y = f2bf(f.y); u.z = f2bf(f.z); u.w = f2bf(f.w);
    out[i] = u;
  }
}

// ---------------------------------------------------------------------------
// Tiled bf16 GEMM: Out[row, col] = (sum_k A[row,k] * W[col,k] + bias[col]) * scale
// Block tile 128x128, K-tile 32, double-buffered async-to-LDS pipeline.
// 8 waves in 4(M) x 2(N); each wave owns a 32x64 tile = 8 WMMA accumulators.
// MODE 0/1: store bf16 natural (Q / K)
// MODE 2  : store bf16 transposed per-head -> Vt[b,h,d,t]  (T=2048,H=16,D=128)
// MODE 3  : store f32 natural (final output projection)
// ---------------------------------------------------------------------------
template <int MODE>
__global__ __launch_bounds__(256)
void gemm_bf16(const unsigned short* __restrict__ A,
               const unsigned short* __restrict__ W,
               const float* __restrict__ bias,
               void* __restrict__ Out,
               int Ndim, int Kdim, float scale) {
  constexpr int BM = 128, BN = 128, BK = 32;
  __shared__ __align__(128) unsigned short As[2][BM * BK];
  __shared__ __align__(128) unsigned short Bs[2][BN * BK];

  const int tid  = threadIdx.x;
  const int lane = tid & 31;
  const int wave = tid >> 5;
  const int hp   = lane >> 4;   // lane-half 0/1
  const int l16  = lane & 15;
  const int wm   = wave >> 1;   // 0..3
  const int wn   = wave & 1;    // 0..1

  const int mBase = blockIdx.y * BM;
  const int nBase = blockIdx.x * BN;

  const v8f vz = {0.f, 0.f, 0.f, 0.f, 0.f, 0.f, 0.f, 0.f};
  v8f acc[2][4];
#pragma unroll
  for (int mi = 0; mi < 2; ++mi)
#pragma unroll
    for (int ni = 0; ni < 4; ++ni) acc[mi][ni] = vz;

  // staging: 256 threads, 2 threads/row, 16 elems (32B = 2x b128) each
  const int sRow = tid >> 1;
  const int sOff = (tid & 1) * 16;
  const unsigned short* gA = A + (size_t)(mBase + sRow) * Kdim + sOff;
  const unsigned short* gB = W + (size_t)(nBase + sRow) * Kdim + sOff;
  const unsigned aoff = lds_off(&As[0][sRow * BK + sOff]);
  const unsigned boff = lds_off(&Bs[0][sRow * BK + sOff]);
  constexpr unsigned bufStride = BM * BK * 2;  // 8 KB per buffer

  auto stage = [&](int buf, int k0) {
    const unsigned bo = (unsigned)buf * bufStride;
    async_load_b128(aoff + bo,      gA + k0);
    async_load_b128(aoff + bo + 16, gA + k0 + 8);
    async_load_b128(boff + bo,      gB + k0);
    async_load_b128(boff + bo + 16, gB + k0 + 8);
  };

  stage(0, 0);

  int it = 0;
  for (int k0 = 0; k0 < Kdim; k0 += BK, ++it) {
    const int cur = it & 1;
    wait_async0();       // our async DMA for tile `cur` complete
    __syncthreads();     // everyone's DMA complete + prev reads retired
    if (k0 + BK < Kdim) stage(cur ^ 1, k0 + BK);  // overlap next DMA w/ WMMA

    // A fragments: 16x32 bf16 (one contiguous 32B K-chunk per lane)
    v16bf afr[2];
#pragma unroll
    for (int mi = 0; mi < 2; ++mi) {
      const unsigned short* p = &As[cur][(wm * 32 + mi * 16 + l16) * BK + hp * 16];
      afr[mi] = *(const v16bf*)p;
    }
    // B fragments: 32x16 bf16 (one contiguous 32B chunk per lane)
    v16bf bfr[4];
#pragma unroll
    for (int ni = 0; ni < 4; ++ni)
      bfr[ni] = *(const v16bf*)&Bs[cur][(wn * 64 + ni * 16 + l16) * BK + hp * 16];

#pragma unroll
    for (int mi = 0; mi < 2; ++mi)
#pragma unroll
      for (int ni = 0; ni < 4; ++ni)
        acc[mi][ni] = WMMA_BF16(afr[mi], bfr[ni], acc[mi][ni]);
  }

  // epilogue (C layout: VGPR r -> row r + 8*half, col = lane%16)
#pragma unroll
  for (int mi = 0; mi < 2; ++mi) {
#pragma unroll
    for (int ni = 0; ni < 4; ++ni) {
      const int col = nBase + wn * 64 + ni * 16 + l16;
      const float bv = bias[col];
#pragma unroll
      for (int r = 0; r < 8; ++r) {
        const int row = mBase + wm * 32 + mi * 16 + hp * 8 + r;
        const float v = (acc[mi][ni][r] + bv) * scale;
        if constexpr (MODE == 2) {
          const int b = row >> 11, t = row & 2047;     // T = 2048
          const int h = col >> 7,  d = col & 127;      // D = 128
          ((unsigned short*)Out)[(((size_t)((b << 4) + h) * 128 + d) << 11) + t] = f2bf(v);
        } else if constexpr (MODE == 3) {
          ((float*)Out)[(size_t)row * Ndim + col] = v;
        } else {
          ((unsigned short*)Out)[(size_t)row * Ndim + col] = f2bf(v);
        }
      }
    }
  }
}

// ---------------------------------------------------------------------------
// Flash attention with ALiBi. One block = one (b,h) and 128 query rows.
// 8 waves x 16 rows each; j streamed in double-buffered 32-wide tiles via
// async-to-LDS DMA; online softmax; WMMA for both Q.K^T and P.V.
// Q/K natural bf16 [b,t,h*128+d]; V transposed bf16 [b,h,d,t].
// ---------------------------------------------------------------------------
__global__ __launch_bounds__(256)
void attn_flash(const unsigned short* __restrict__ Q,
                const unsigned short* __restrict__ Kg,
                const unsigned short* __restrict__ Vt,
                const float* __restrict__ slopes,
                unsigned short* __restrict__ Og) {
  constexpr int Tn = 2048, E = 2048, Dh = 128;
  __shared__ __align__(128) unsigned short Ks[2][32 * Dh];   // [j][d]
  __shared__ __align__(128) unsigned short Vs[2][Dh * 32];   // [d][j]
  __shared__ __align__(128) unsigned short Ps[8 * 16 * 32];  // per-wave P tile

  const int tid = threadIdx.x, lane = tid & 31, wave = tid >> 5;
  const int hp = lane >> 4, l16 = lane & 15;
  const int bh = blockIdx.y;            // b*H + h
  const int b = bh >> 4, h = bh & 15;
  const int iBase = blockIdx.x * 128;
  const float slope = slopes[h];

  // load Q fragments once: rows iBase + wave*16 + 0..15, d = 0..127
  v16bf qf[4];
  {
    const unsigned short* qrow =
        Q + ((size_t)(b * Tn + iBase + wave * 16 + l16)) * E + h * Dh;
#pragma unroll
    for (int dk = 0; dk < 4; ++dk) {
      const unsigned short* p = qrow + dk * 32 + hp * 16;
      qf[dk] = *(const v16bf*)p;
    }
  }

  const v8f vz = {0.f, 0.f, 0.f, 0.f, 0.f, 0.f, 0.f, 0.f};
  v8f o[8];
#pragma unroll
  for (int ni = 0; ni < 8; ++ni) o[ni] = vz;
  float m[8], s[8];
#pragma unroll
  for (int r = 0; r < 8; ++r) { m[r] = -3.0e38f; s[r] = 0.f; }

  // async staging setup: K tile 32x128 (2x b128/thread), V tile 128x32 (2x b128)
  const int kRow = tid >> 3, kOff = (tid & 7) * 16;
  const int vRow = tid >> 1, vOff = (tid & 1) * 16;
  const unsigned short* gK = Kg + (size_t)b * Tn * E + h * Dh + (size_t)kRow * E + kOff;
  const unsigned short* gV = Vt + (size_t)bh * Dh * Tn + (size_t)vRow * Tn + vOff;
  const unsigned kloff = lds_off(&Ks[0][kRow * Dh + kOff]);
  const unsigned vloff = lds_off(&Vs[0][vRow * 32 + vOff]);
  constexpr unsigned kStride = 32 * Dh * 2;  // 8 KB
  constexpr unsigned vStride = Dh * 32 * 2;  // 8 KB

  auto stageKV = [&](int buf, int j0) {
    const unsigned kb = (unsigned)buf * kStride;
    const unsigned vb = (unsigned)buf * vStride;
    async_load_b128(kloff + kb,      gK + (size_t)j0 * E);
    async_load_b128(kloff + kb + 16, gK + (size_t)j0 * E + 8);
    async_load_b128(vloff + vb,      gV + j0);
    async_load_b128(vloff + vb + 16, gV + j0 + 8);
  };

  stageKV(0, 0);

  int it = 0;
  for (int j0 = 0; j0 < Tn; j0 += 32, ++it) {
    const int cur = it & 1;
    wait_async0();
    __syncthreads();
    if (j0 + 32 < Tn) stageKV(cur ^ 1, j0 + 32);  // overlap next tile DMA

    // scores S = Q . K^T  (two 16x16 j-tiles, K-loop over d)
    v8f S0 = vz, S1 = vz;
#pragma unroll
    for (int dk = 0; dk < 4; ++dk) {
      v16bf k0f = *(const v16bf*)&Ks[cur][l16 * Dh + dk * 32 + hp * 16];
      v16bf k1f = *(const v16bf*)&Ks[cur][(16 + l16) * Dh + dk * 32 + hp * 16];
      S0 = WMMA_BF16(qf[dk], k0f, S0);
      S1 = WMMA_BF16(qf[dk], k1f, S1);
    }

    // ALiBi bias + online softmax (per output row r)
#pragma unroll
    for (int r = 0; r < 8; ++r) {
      const float fi = (float)(iBase + wave * 16 + hp * 8 + r);
      float sc0 = S0[r] - slope * fabsf((float)(j0 + l16) - fi);
      float sc1 = S1[r] - slope * fabsf((float)(j0 + 16 + l16) - fi);
      float mx = red_max16(fmaxf(sc0, sc1));
      mx = fmaxf(mx, m[r]);
      const float fac = __expf(m[r] - mx);
      const float p0 = __expf(sc0 - mx);
      const float p1 = __expf(sc1 - mx);
      s[r] = s[r] * fac + red_sum16(p0 + p1);
      m[r] = mx;
#pragma unroll
      for (int ni = 0; ni < 8; ++ni) o[ni][r] *= fac;
      // scatter P (C layout) into LDS as a row-major 16x32 tile
      unsigned short* pp = &Ps[(wave * 16 + hp * 8 + r) * 32];
      pp[l16]      = f2bf(p0);
      pp[16 + l16] = f2bf(p1);
    }
    __syncthreads();

    // P.V : A = P (16 x 32 over j), B = V (32 j x 16 d), 8 d-subtiles
    v16bf pf;
    {
      const unsigned short* p = &Ps[(wave * 16 + l16) * 32 + hp * 16];
      pf = *(const v16bf*)p;
    }
#pragma unroll
    for (int ni = 0; ni < 8; ++ni) {
      v16bf vf = *(const v16bf*)&Vs[cur][(ni * 16 + l16) * 32 + hp * 16];
      o[ni] = WMMA_BF16(pf, vf, o[ni]);
    }
  }

  // normalize and store bf16 attention output [b,t,h*128+d]
  const size_t orow = ((size_t)(b * Tn + iBase + wave * 16)) * E + h * Dh;
#pragma unroll
  for (int ni = 0; ni < 8; ++ni) {
#pragma unroll
    for (int r = 0; r < 8; ++r) {
      const float v = o[ni][r] / s[r];
      Og[orow + (size_t)(hp * 8 + r) * E + ni * 16 + l16] = f2bf(v);
    }
  }
}

// ---------------------------------------------------------------------------
// Host-side orchestration
// ---------------------------------------------------------------------------
extern "C" void kernel_launch(void* const* d_in, const int* in_sizes, int n_in,
                              void* d_out, int out_size, void* d_ws, size_t ws_size,
                              hipStream_t stream) {
  (void)in_sizes; (void)n_in; (void)out_size; (void)ws_size;
  const float* x      = (const float*)d_in[0];
  const float* Wq     = (const float*)d_in[1];
  const float* bq     = (const float*)d_in[2];
  const float* Wk     = (const float*)d_in[3];
  const float* bk     = (const float*)d_in[4];
  const float* Wv     = (const float*)d_in[5];
  const float* bv     = (const float*)d_in[6];
  const float* Wo     = (const float*)d_in[7];
  const float* bo     = (const float*)d_in[8];
  const float* slopes = (const float*)d_in[9];

  constexpr int Bb = 2, Tn = 2048, E = 2048, H = 16;
  constexpr size_t MT = (size_t)Bb * Tn;   // 4096 token rows
  constexpr size_t XE = MT * E;            // 8M elements
  constexpr size_t EE = (size_t)E * E;     // 4M elements

  unsigned char* ws = (unsigned char*)d_ws;
  size_t off = 0;
  auto carve = [&](size_t bytes) -> void* {
    void* p = ws + off;
    off += (bytes + 255) & ~(size_t)255;
    return p;
  };
  unsigned short* x_bf  = (unsigned short*)carve(XE * 2);
  unsigned short* wq_bf = (unsigned short*)carve(EE * 2);
  unsigned short* wk_bf = (unsigned short*)carve(EE * 2);
  unsigned short* wv_bf = (unsigned short*)carve(EE * 2);
  unsigned short* wo_bf = (unsigned short*)carve(EE * 2);
  unsigned short* q_bf  = (unsigned short*)carve(XE * 2);
  unsigned short* k_bf  = (unsigned short*)carve(XE * 2);
  unsigned short* vt_bf = (unsigned short*)carve(XE * 2);
  unsigned short* ao_bf = (unsigned short*)carve(XE * 2);

  cvt_f32_bf16<<<1024, 256, 0, stream>>>((const float4*)x,  (ushort4*)x_bf,  (int)(XE / 4));
  cvt_f32_bf16<<<1024, 256, 0, stream>>>((const float4*)Wq, (ushort4*)wq_bf, (int)(EE / 4));
  cvt_f32_bf16<<<1024, 256, 0, stream>>>((const float4*)Wk, (ushort4*)wk_bf, (int)(EE / 4));
  cvt_f32_bf16<<<1024, 256, 0, stream>>>((const float4*)Wv, (ushort4*)wv_bf, (int)(EE / 4));
  cvt_f32_bf16<<<1024, 256, 0, stream>>>((const float4*)Wo, (ushort4*)wo_bf, (int)(EE / 4));

  dim3 gg(E / 128, (int)(MT / 128));  // (16, 32)
  const float scale = 1.0f / sqrtf(128.0f);
  gemm_bf16<0><<<gg, 256, 0, stream>>>(x_bf, wq_bf, bq, q_bf,  E, E, scale);
  gemm_bf16<1><<<gg, 256, 0, stream>>>(x_bf, wk_bf, bk, k_bf,  E, E, 1.0f);
  gemm_bf16<2><<<gg, 256, 0, stream>>>(x_bf, wv_bf, bv, vt_bf, E, E, 1.0f);

  dim3 ga(Tn / 128, Bb * H);          // (16, 32)
  attn_flash<<<ga, 256, 0, stream>>>(q_bf, k_bf, vt_bf, slopes, ao_bf);

  gemm_bf16<3><<<gg, 256, 0, stream>>>(ao_bf, wo_bf, bo, d_out, E, E, 1.0f);
}
